// FSCModule_34119220199905
// MI455X (gfx1250) — compile-verified, matrix operand order checked by reading, not run on hardware
//
#include <hip/hip_runtime.h>
#include <math.h>

// ---------------- problem constants ----------------
#define D_MODEL 96
#define D_STATE 16
#define D_INNER 192
#define DT_RANK 6
#define KDIR    4
#define NB      4
#define HH      64
#define WW      64
#define LL      4096
#define C_DBL   38

typedef __attribute__((ext_vector_type(16))) _Float16 v16h;
typedef __attribute__((ext_vector_type(8)))  float    v8f;

// fragment element e (0..15) -> K offset within a 32-wide K chunk (CDNA5 16-bit layout)
__device__ __forceinline__ int kmap(int e, int half) {
    return (e & 7) + ((e >> 3) << 4) + (half << 3);
}
__device__ __forceinline__ float siluf(float v) { return v / (1.0f + __expf(-v)); }

// inverse of kmap: r = K&31 -> (half, e)
__device__ __forceinline__ void kinv(int r, int& half, int& e) {
    half = (r >> 3) & 1;
    e    = (r & 7) | ((r >> 4) << 3);
}

// panel sizes (in _Float16 elements)
#define XA16_SZ   ((size_t)1024 * 3 * 32 * 16)   // x panel:   1024 mtiles x 3 kt
#define WI16_SZ   ((size_t)24 * 3 * 32 * 16)     // in_proj_w: 24 ntiles x 3 kt
#define WP16_SZ   ((size_t)4 * 3 * 6 * 32 * 16)  // x_proj_w:  4 dirs x 3 ntiles x 6 kt
#define WO16_SZ   ((size_t)6 * 6 * 32 * 16)      // out_proj_w:6 ntiles x 6 kt
#define XSDIR_SZ  ((size_t)1024 * 6 * 32 * 16)   // per-direction xs panel
#define YA16_SZ   ((size_t)1024 * 6 * 32 * 16)   // merged-y panel for out_proj

// ---------------------------------------------------------------
// Prep: swizzle the three weight matrices into f16 fragment panels
// slots: [0,72) in_proj, [72,144) x_proj, [144,180) out_proj
// ---------------------------------------------------------------
__global__ void k_prep_w(const float* __restrict__ wi, const float* __restrict__ wp,
                         const float* __restrict__ wo,
                         _Float16* __restrict__ wi16, _Float16* __restrict__ wp16,
                         _Float16* __restrict__ wo16) {
    int t = blockIdx.x * blockDim.x + threadIdx.x;
    if (t >= 180 * 32) return;
    int lane = t & 31, half = lane >> 4, idx = lane & 15;
    int slot = t >> 5;
    v16h frag;
    if (slot < 72) {                       // in_proj: (384,96)
        int nt = slot / 3, kt = slot % 3;
#pragma unroll
        for (int e = 0; e < 16; ++e)
            frag[e] = (_Float16)wi[(nt * 16 + idx) * D_MODEL + kt * 32 + kmap(e, half)];
        *(v16h*)(wi16 + (size_t)t * 16) = frag;
    } else if (slot < 144) {               // x_proj: (4,38,192), N padded to 48
        int s = slot - 72;                 // s = (kdir*3 + nt)*6 + kt
        int kt = s % 6, nt = (s / 6) % 3, kd = s / 18;
        int c = nt * 16 + idx; if (c >= C_DBL) c = C_DBL - 1;
#pragma unroll
        for (int e = 0; e < 16; ++e)
            frag[e] = (_Float16)wp[((size_t)(kd * C_DBL + c)) * D_INNER + kt * 32 + kmap(e, half)];
        *(v16h*)(wp16 + (size_t)(s * 32 + lane) * 16) = frag;
    } else {                               // out_proj: (96,192)
        int s = slot - 144;                // s = nt*6 + kt
        int nt = s / 6, kt = s % 6;
#pragma unroll
        for (int e = 0; e < 16; ++e)
            frag[e] = (_Float16)wo[(size_t)(nt * 16 + idx) * D_INNER + kt * 32 + kmap(e, half)];
        *(v16h*)(wo16 + (size_t)(s * 32 + lane) * 16) = frag;
    }
}

// ---------------------------------------------------------------
// Prep: swizzle x (16384 x 96) into f16 fragment panel
// ---------------------------------------------------------------
__global__ void k_prep_x(const float* __restrict__ x, _Float16* __restrict__ xa16) {
    int t = blockIdx.x * blockDim.x + threadIdx.x;    // 1024*3*32 threads
    if (t >= 1024 * 3 * 32) return;
    int lane = t & 31, half = lane >> 4, idx = lane & 15;
    int slot = t >> 5;
    int mt = slot / 3, kt = slot % 3;
    v16h frag;
#pragma unroll
    for (int e = 0; e < 16; ++e)
        frag[e] = (_Float16)x[(size_t)(mt * 16 + idx) * D_MODEL + kt * 32 + kmap(e, half)];
    *(v16h*)(xa16 + (size_t)t * 16) = frag;
}

// ---------------------------------------------------------------
// Kernel 1: xz = x @ in_proj_w.T : pure fragment loads + WMMA
// ---------------------------------------------------------------
__global__ void k_in_proj(const _Float16* __restrict__ xa16,
                          const _Float16* __restrict__ wi16,
                          float* __restrict__ xz) {
    const int mt = blockIdx.x, nt = blockIdx.y;
    const int lane = threadIdx.x & 31;
    const int half = lane >> 4, idx = lane & 15;
    const v16h* A = (const v16h*)xa16;
    const v16h* B = (const v16h*)wi16;
    v8f acc = {};
#pragma unroll
    for (int kt = 0; kt < 3; ++kt) {
        v16h a = A[(size_t)(mt * 3 + kt) * 32 + lane];
        v16h b = B[(size_t)(nt * 3 + kt) * 32 + lane];
        acc = __builtin_amdgcn_wmma_f32_16x16x32_f16(false, a, false, b,
                                                     (short)0, acc, false, false);
    }
#pragma unroll
    for (int r = 0; r < 8; ++r)
        xz[(size_t)(mt * 16 + r + half * 8) * (2 * D_INNER) + (nt * 16 + idx)] = acc[r];
}

// ---------------------------------------------------------------
// Kernel 2: depthwise 3x3 + bias + SiLU; writes fp32 planar xi and
// scatters f16 into the 4 direction-gathered A panels for x_dbl
// ---------------------------------------------------------------
__global__ void k_dwconv(const float* __restrict__ xz,
                         const float* __restrict__ cw, const float* __restrict__ cb,
                         float* __restrict__ xi, _Float16* __restrict__ xs16) {
    int t = blockIdx.x * blockDim.x + threadIdx.x;
    if (t >= NB * D_INNER * LL) return;
    int w = t & 63, h = (t >> 6) & 63;
    int d = (t >> 12) % D_INNER;
    int b = t / (LL * D_INNER);

    float s = cb[d];
#pragma unroll
    for (int dh = -1; dh <= 1; ++dh)
#pragma unroll
        for (int dw = -1; dw <= 1; ++dw) {
            int hh = h + dh, ww = w + dw;
            if (hh >= 0 && hh < HH && ww >= 0 && ww < WW)
                s += xz[((size_t)(b * HH + hh) * WW + ww) * (2 * D_INNER) + d] *
                     cw[d * 9 + (dh + 1) * 3 + (dw + 1)];
        }
    float v = siluf(s);
    int l = h * WW + w;
    xi[((size_t)(b * D_INNER + d)) * LL + l] = v;

    // scatter into fragment panels: scan position per direction s.t. gather hits l
    _Float16 hv = (_Float16)v;
    int lt = ((l & 63) << 6) | (l >> 6);
    int lk0 = l, lk1 = lt, lk2 = (LL - 1) - l, lk3 = (LL - 1) - lt;
    int kt = d >> 5, r = d & 31, hf, e;
    kinv(r, hf, e);
#pragma unroll
    for (int kd = 0; kd < 4; ++kd) {
        int lk = (kd == 0) ? lk0 : (kd == 1) ? lk1 : (kd == 2) ? lk2 : lk3;
        int m = b * LL + lk;
        int mt = m >> 4, idx = m & 15;
        xs16[(size_t)kd * XSDIR_SZ +
             (((size_t)mt * 6 + kt) * 32 + (hf * 16 + idx)) * 16 + e] = hv;
    }
}

// ---------------------------------------------------------------
// Kernel 3: x_dbl GEMM — fragment loads + WMMA, K=192 (6 steps)
// ---------------------------------------------------------------
__global__ void k_xdbl(const _Float16* __restrict__ xs16,
                       const _Float16* __restrict__ wp16,
                       float* __restrict__ xdbl) {
    const int mt = blockIdx.x, nt = blockIdx.y, kdir = blockIdx.z;
    const int lane = threadIdx.x & 31;
    const int half = lane >> 4, idx = lane & 15;
    const v16h* A = (const v16h*)(xs16 + (size_t)kdir * XSDIR_SZ);
    const v16h* B = (const v16h*)wp16;
    v8f acc = {};
#pragma unroll
    for (int kt = 0; kt < 6; ++kt) {
        v16h a = A[((size_t)mt * 6 + kt) * 32 + lane];
        v16h b = B[((size_t)((kdir * 3 + nt) * 6 + kt)) * 32 + lane];
        acc = __builtin_amdgcn_wmma_f32_16x16x32_f16(false, a, false, b,
                                                     (short)0, acc, false, false);
    }
    const int c = nt * 16 + idx;
#pragma unroll
    for (int r = 0; r < 8; ++r) {
        int m = mt * 16 + r + half * 8;
        int b2 = m >> 12, l = m & (LL - 1);
        if (c < C_DBL)
            xdbl[(((size_t)(b2 * KDIR + kdir) * C_DBL) + c) * LL + l] = acc[r];
    }
}

// ---------------------------------------------------------------
// Kernel 4: delta = softplus(dt_proj(dts)+b). One block per (b,k,l-chunk);
// dts slab staged to LDS via gfx1250 async-to-LDS, then swept over d.
// ---------------------------------------------------------------
__global__ void k_delta(const float* __restrict__ xdbl,
                        const float* __restrict__ dtw, const float* __restrict__ dtb,
                        float* __restrict__ dlt) {
    int bk = blockIdx.x >> 4;          // b*4+k
    int l0 = (blockIdx.x & 15) * 256;
    int tid = threadIdx.x;
    int k = bk & 3;
    __shared__ float sdts[DT_RANK][256];

    const float* src = xdbl + ((size_t)bk * C_DBL) * LL + l0 + tid;
#pragma unroll
    for (int r = 0; r < DT_RANK; ++r) {
        unsigned lds = (unsigned)(size_t)&sdts[r][tid];
        unsigned long long ga = (unsigned long long)(size_t)(src + (size_t)r * LL);
        asm volatile("global_load_async_to_lds_b32 %0, %1, off"
                     :: "v"(lds), "v"(ga) : "memory");
    }
    asm volatile("s_wait_asynccnt 0" ::: "memory");
    __syncthreads();

    float* dst = dlt + ((size_t)bk * D_INNER) * LL + l0 + tid;
    const float* wk = dtw + (size_t)k * D_INNER * DT_RANK;
    const float* bb = dtb + (size_t)k * D_INNER;
    for (int d = 0; d < D_INNER; ++d) {
        float s = bb[d];
#pragma unroll
        for (int r = 0; r < DT_RANK; ++r)
            s += sdts[r][tid] * wk[d * DT_RANK + r];
        s = (s > 20.0f) ? s : log1pf(__expf(s));
        dst[(size_t)d * LL] = s;
    }
}

// ---------------------------------------------------------------
// Kernel 5: selective scan. Wave = 2 channels, lanes = state n.
// 16-step chunks: coalesced preloads + shfl broadcast, registered B/C.
// ---------------------------------------------------------------
__global__ void k_scan(const float* __restrict__ xi,
                       const float* __restrict__ xdbl,
                       const float* __restrict__ dlt,
                       const float* __restrict__ A_logs,
                       const float* __restrict__ Ds,
                       float* __restrict__ oy) {
    int wave = blockIdx.x * (blockDim.x >> 5) + (threadIdx.x >> 5);
    int lane = threadIdx.x & 31;
    int half = lane >> 4, n = lane & 15;
    int dp = wave % (D_INNER / 2);
    int k  = (wave / (D_INNER / 2)) & 3;
    int b  = wave / (KDIR * D_INNER / 2);
    int d  = dp * 2 + half;

    float An = -__expf(A_logs[(size_t)(k * D_INNER + d) * D_STATE + n]);
    float Dv = Ds[k * D_INNER + d];

    const float* dptr = dlt + ((size_t)((b * KDIR + k) * D_INNER + d)) * LL;
    const float* uptr = xi  + ((size_t)(b * D_INNER + d)) * LL;
    const float* Bp = xdbl + ((size_t)((b * KDIR + k) * C_DBL) + DT_RANK + n) * LL;
    const float* Cp = xdbl + ((size_t)((b * KDIR + k) * C_DBL) + DT_RANK + D_STATE + n) * LL;
    float* yp = oy + ((size_t)((b * KDIR + k) * D_INNER + d)) * LL;

    float h = 0.0f;
    for (int l0 = 0; l0 < LL; l0 += 16) {
        // coalesced per-lane chunk preloads
        int lpre = l0 + n;
        int off  = (k == 0) ? lpre
                 : (k == 1) ? (((lpre & 63) << 6) | (lpre >> 6))
                 : (k == 2) ? ((LL - 1) - lpre)
                 : ((((((LL - 1) - lpre) & 63) << 6) | (((LL - 1) - lpre) >> 6)));
        float dch = dptr[lpre];
        float uch = uptr[off];
        float bb[16], cc[16];
#pragma unroll
        for (int i = 0; i < 4; ++i) {
            *(float4*)(bb + 4 * i) = *(const float4*)(Bp + l0 + 4 * i);
            *(float4*)(cc + 4 * i) = *(const float4*)(Cp + l0 + 4 * i);
        }
        __builtin_prefetch(dptr + l0 + 128, 0, 0);
        __builtin_prefetch(Bp + l0 + 128, 0, 0);
        __builtin_prefetch(Cp + l0 + 128, 0, 0);

        float yv = 0.0f;
#pragma unroll
        for (int j = 0; j < 16; ++j) {
            float dt = __shfl(dch, half * 16 + j, 32);
            float u  = __shfl(uch, half * 16 + j, 32);
            float dA = __expf(dt * An);
            h = __fmaf_rn(dA, h, dt * u * bb[j]);
            float p = h * cc[j];
            p += __shfl_xor(p, 1, 32);
            p += __shfl_xor(p, 2, 32);
            p += __shfl_xor(p, 4, 32);
            p += __shfl_xor(p, 8, 32);
            if (j == n) yv = p + Dv * u;
        }
        yp[l0 + n] = yv;   // coalesced
    }
}

// ---------------------------------------------------------------
// Kernel 6: merge 4 directions + LayerNorm + SiLU(z) gate ->
// write straight into the f16 fragment panel for out_proj
// ---------------------------------------------------------------
__global__ void k_merge_ln(const float* __restrict__ oy,
                           const float* __restrict__ xz,
                           const float* __restrict__ g, const float* __restrict__ bn,
                           _Float16* __restrict__ ya16) {
    int m = blockIdx.x;                 // b*4096 + l
    int b = m >> 12, l = m & (LL - 1);
    int d = threadIdx.x;
    int t1 = ((l & 63) << 6) | (l >> 6);

    size_t base = (size_t)(b * KDIR) * D_INNER * LL + (size_t)d * LL;
    float y = oy[base + 0 * (size_t)D_INNER * LL + l]
            + oy[base + 2 * (size_t)D_INNER * LL + (LL - 1 - l)]
            + oy[base + 1 * (size_t)D_INNER * LL + t1]
            + oy[base + 3 * (size_t)D_INNER * LL + (LL - 1 - t1)];

    float s = y, s2 = y * y;
#pragma unroll
    for (int o = 16; o; o >>= 1) {
        s  += __shfl_xor(s, o, 32);
        s2 += __shfl_xor(s2, o, 32);
    }
    __shared__ float sh[12];
    int wid = threadIdx.x >> 5;
    if ((threadIdx.x & 31) == 0) { sh[wid] = s; sh[6 + wid] = s2; }
    __syncthreads();
    if (threadIdx.x == 0) {
        float a = 0.f, a2 = 0.f;
#pragma unroll
        for (int i = 0; i < 6; ++i) { a += sh[i]; a2 += sh[6 + i]; }
        sh[0] = a; sh[6] = a2;
    }
    __syncthreads();
    float mu  = sh[0] * (1.0f / D_INNER);
    float var = sh[6] * (1.0f / D_INNER) - mu * mu;
    float ny  = (y - mu) * rsqrtf(var + 1e-5f) * g[d] + bn[d];
    float z   = xz[((size_t)m) * (2 * D_INNER) + D_INNER + d];
    float val = ny * siluf(z);

    int mt = m >> 4, idx = m & 15;
    int kt = d >> 5, r = d & 31, hf, e;
    kinv(r, hf, e);
    ya16[(((size_t)mt * 6 + kt) * 32 + (hf * 16 + idx)) * 16 + e] = (_Float16)val;
}

// ---------------------------------------------------------------
// Kernel 7: out = y_ln @ out_proj_w.T — fragment loads + WMMA
// ---------------------------------------------------------------
__global__ void k_out_proj(const _Float16* __restrict__ ya16,
                           const _Float16* __restrict__ wo16,
                           float* __restrict__ out) {
    const int mt = blockIdx.x, nt = blockIdx.y;
    const int lane = threadIdx.x & 31;
    const int half = lane >> 4, idx = lane & 15;
    const v16h* A = (const v16h*)ya16;
    const v16h* B = (const v16h*)wo16;
    v8f acc = {};
#pragma unroll
    for (int kt = 0; kt < 6; ++kt) {
        v16h a = A[((size_t)mt * 6 + kt) * 32 + lane];
        v16h b = B[((size_t)(nt * 6 + kt)) * 32 + lane];
        acc = __builtin_amdgcn_wmma_f32_16x16x32_f16(false, a, false, b,
                                                     (short)0, acc, false, false);
    }
#pragma unroll
    for (int r = 0; r < 8; ++r)
        out[(size_t)(mt * 16 + r + half * 8) * D_MODEL + (nt * 16 + idx)] = acc[r];
}

// ---------------------------------------------------------------
extern "C" void kernel_launch(void* const* d_in, const int* in_sizes, int n_in,
                              void* d_out, int out_size, void* d_ws, size_t ws_size,
                              hipStream_t stream) {
    const float* x         = (const float*)d_in[0];
    const float* in_proj_w = (const float*)d_in[1];
    const float* conv_w    = (const float*)d_in[2];
    const float* conv_b    = (const float*)d_in[3];
    const float* x_proj_w  = (const float*)d_in[4];
    const float* dt_proj_w = (const float*)d_in[5];
    const float* dt_proj_b = (const float*)d_in[6];
    const float* A_logs    = (const float*)d_in[7];
    const float* Ds        = (const float*)d_in[8];
    const float* norm_g    = (const float*)d_in[9];
    const float* norm_b    = (const float*)d_in[10];
    const float* out_proj_w= (const float*)d_in[11];

    float* ws   = (float*)d_ws;
    float* xz   = ws;                                      // 16384*384
    float* xi   = xz   + (size_t)16384 * 384;              // 4*192*4096
    float* xdbl = xi   + (size_t)NB * D_INNER * LL;        // 4*4*38*4096
    float* dlt  = xdbl + (size_t)NB * KDIR * C_DBL * LL;   // 4*4*192*4096
    float* oy   = dlt  + (size_t)NB * KDIR * D_INNER * LL; // 4*4*192*4096
    _Float16* xa16 = (_Float16*)(oy + (size_t)NB * KDIR * D_INNER * LL);
    _Float16* wi16 = xa16 + XA16_SZ;
    _Float16* wp16 = wi16 + WI16_SZ;
    _Float16* wo16 = wp16 + WP16_SZ;
    _Float16* xs16 = wo16 + WO16_SZ;                       // 4 * XSDIR_SZ
    _Float16* ya16 = xs16 + 4 * XSDIR_SZ;

    // 0. swizzle weights + x into f16 WMMA fragment panels
    k_prep_w<<<(180 * 32 + 127) / 128, 128, 0, stream>>>(in_proj_w, x_proj_w, out_proj_w,
                                                         wi16, wp16, wo16);
    k_prep_x<<<(1024 * 3 * 32) / 256, 256, 0, stream>>>(x, xa16);
    // 1. in_proj GEMM (WMMA)
    k_in_proj<<<dim3(1024, 24), 32, 0, stream>>>(xa16, wi16, xz);
    // 2. depthwise conv + SiLU (+ scatter into direction panels)
    k_dwconv<<<(NB * D_INNER * LL + 255) / 256, 256, 0, stream>>>(xz, conv_w, conv_b, xi, xs16);
    // 3. x_dbl GEMM (WMMA)
    k_xdbl<<<dim3(1024, 3, KDIR), 32, 0, stream>>>(xs16, wp16, xdbl);
    // 4. delta (async-to-LDS staging)
    k_delta<<<NB * KDIR * 16, 256, 0, stream>>>(xdbl, dt_proj_w, dt_proj_b, dlt);
    // 5. selective scan
    k_scan<<<(NB * KDIR * D_INNER / 2) / 2, 64, 0, stream>>>(xi, xdbl, dlt, A_logs, Ds, oy);
    // 6. merge + LayerNorm + gate -> fragment panel
    k_merge_ln<<<NB * LL, D_INNER, 0, stream>>>(oy, xz, norm_g, norm_b, ya16);
    // 7. out_proj GEMM (WMMA)
    k_out_proj<<<dim3(1024, 6), 32, 0, stream>>>(ya16, wo16, (float*)d_out);
}
